// Lfm2ShortConv_32890859553428
// MI455X (gfx1250) — matrix operand order repaired and checked by reading, not compiled, not run on hardware
//
#include <hip/hip_runtime.h>

// ---------------------------------------------------------------------------
// LFM2 short-conv block for gfx1250 (MI455X), bf16 WMMA + async-to-LDS path.
//   out        = (c * depthwise_conv(b*x)) @ w_out^T
//   new_states = last K-1 rows of (b*x) per sequence
// ---------------------------------------------------------------------------

#define N_TOK   16384      // NUM_SEQS * SEQ_LEN
#define SEQLEN  4096
#define HID     2048
#define DCONV   2048
#define KCONV   4

typedef __attribute__((ext_vector_type(16))) __bf16 v16bf;
typedef __attribute__((ext_vector_type(8)))  float  v8f;

union FragBF16 {
  v16bf v;
  uint4 q[2];
};

__device__ __forceinline__ unsigned short f2bf(float x) {
  unsigned u = __float_as_uint(x);
  u += 0x7FFFu + ((u >> 16) & 1u);          // round-to-nearest-even
  return (unsigned short)(u >> 16);
}
__device__ __forceinline__ float bf2f(unsigned short h) {
  return __uint_as_float(((unsigned)h) << 16);
}

// gfx1250 async copy: global -> LDS, 16B per lane, tracked by ASYNCcnt.
// GV mode: per-lane 64-bit global address, vdst VGPR holds LDS byte address.
__device__ __forceinline__ void async_b128(void* lds_ptr, const void* gptr) {
  asm volatile("global_load_async_to_lds_b128 %0, %1, off"
               :: "v"((unsigned)(uintptr_t)lds_ptr), "v"(gptr)
               : "memory");
}
#define S_WAIT_ASYNC(imm) asm volatile("s_wait_asynccnt " #imm ::: "memory")

// ---------------------------------------------------------------------------
// f32 -> bf16 conversion, 4 elements/thread
// ---------------------------------------------------------------------------
__global__ void cvt4_f32_bf16(const float4* __restrict__ s,
                              ushort4* __restrict__ d, int n4) {
  int i = blockIdx.x * blockDim.x + threadIdx.x;
  int stride = gridDim.x * blockDim.x;
  for (; i < n4; i += stride) {
    float4 v = s[i];
    d[i] = make_ushort4(f2bf(v.x), f2bf(v.y), f2bf(v.z), f2bf(v.w));
  }
}

// ---------------------------------------------------------------------------
// Tiling: block 128(M) x 128(N per matrix), 512 threads = 16 waves (4x4),
// wave tile 32x32 per matrix.  K-stage 64 = two WMMA K=32 sub-steps.
// Double-buffered LDS fed by global_load_async_to_lds_b128.
// ---------------------------------------------------------------------------
#define BM    128
#define BK2   64           // K elements staged per iteration
#define LSTR2 72           // ushort row stride in LDS (144B, conflict-friendly)
#define TILEU (BM * LSTR2) // ushorts per tile buffer (9216 -> 18432 B)

// ---------------------------------------------------------------------------
// GEMM1 (fused): one A tile vs the b/c/x weight tiles of the same column
// block.  Emits bx = b*x (f32), c (bf16), and new_states rows.
// Dynamic LDS: 8 * TILEU ushorts = 147456 B.
//   tile layout in smem (units of TILEU):  A0, A1, B[buf0][0..2], B[buf1][0..2]
// ---------------------------------------------------------------------------
__global__ __launch_bounds__(512, 1)
void gemm1_bcx(const unsigned short* __restrict__ A,    // hidden bf16 [N_TOK x HID]
               const unsigned short* __restrict__ W,    // w_in   bf16 [3*DCONV x HID]
               float* __restrict__ bx,                  // [N_TOK x DCONV] f32
               unsigned short* __restrict__ cg,         // [N_TOK x DCONV] bf16
               float* __restrict__ states)              // [4*3*DCONV] f32
{
  extern __shared__ unsigned short smem[];

  const int tid  = threadIdx.x;
  const int lane = tid & 31;
  const int wave = tid >> 5;
  const int wm   = wave >> 2;           // 0..3
  const int wn   = wave & 3;            // 0..3
  const int m0   = blockIdx.y * BM;
  const int j0   = blockIdx.x * BM;

  const int lrow = tid >> 2;            // 0..127
  const int cc   = (tid & 3) * 16;      // 0,16,32,48 (ushorts)

  const int lhi = lane >> 4;            // half-wave select
  const int lm  = lane & 15;

  v8f accB[2][2], accC[2][2], accX[2][2];
#pragma unroll
  for (int i = 0; i < 2; ++i)
#pragma unroll
    for (int j = 0; j < 2; ++j) {
      accB[i][j] = (v8f)(0.0f);
      accC[i][j] = (v8f)(0.0f);
      accX[i][j] = (v8f)(0.0f);
    }

  const unsigned short* srcA  = A + (size_t)(m0 + lrow) * HID + cc;
  const unsigned short* srcW0 = W + (size_t)(j0 + lrow) * HID + cc;

  // issue one K-stage (8 async b128 per thread: 2 for A, 2 per weight matrix)
  auto issue = [&](int buf, int kk) {
    unsigned short* dA = smem + buf * TILEU + lrow * LSTR2 + cc;
    async_b128(dA,     srcA + kk);
    async_b128(dA + 8, srcA + kk + 8);
#pragma unroll
    for (int mat = 0; mat < 3; ++mat) {
      unsigned short* dB =
          smem + (2 + buf * 3 + mat) * TILEU + lrow * LSTR2 + cc;
      const unsigned short* s = srcW0 + (size_t)mat * DCONV * HID + kk;
      async_b128(dB,     s);
      async_b128(dB + 8, s + 8);
    }
  };

  const int NIT = HID / BK2;            // 32
  issue(0, 0);

  for (int it = 0; it < NIT; ++it) {
    const int buf = it & 1;
    if (it + 1 < NIT) {
      issue(1 - buf, (it + 1) * BK2);
      S_WAIT_ASYNC(0x8);                // tile `it` (8 older ops) has landed
    } else {
      S_WAIT_ASYNC(0x0);
    }
    __syncthreads();

#pragma unroll
    for (int ks = 0; ks < 2; ++ks) {    // two K=32 sub-steps
      const int ksub = ks * 32;
      FragBF16 a[2];
#pragma unroll
      for (int ti = 0; ti < 2; ++ti) {
        const unsigned short* p =
            smem + buf * TILEU + (wm * 32 + ti * 16 + lm) * LSTR2 + ksub;
        a[ti].q[0] = *(const uint4*)(p + 8 * lhi);
        a[ti].q[1] = *(const uint4*)(p + 8 * lhi + 16);
      }
      FragBF16 b[3][2];
#pragma unroll
      for (int mat = 0; mat < 3; ++mat)
#pragma unroll
        for (int tj = 0; tj < 2; ++tj) {
          const unsigned short* p =
              smem + (2 + buf * 3 + mat) * TILEU +
              (wn * 32 + tj * 16 + lm) * LSTR2 + ksub;
          b[mat][tj].q[0] = *(const uint4*)(p + 8 * lhi);
          b[mat][tj].q[1] = *(const uint4*)(p + 8 * lhi + 16);
        }
#pragma unroll
      for (int ti = 0; ti < 2; ++ti)
#pragma unroll
        for (int tj = 0; tj < 2; ++tj) {
          accB[ti][tj] = __builtin_amdgcn_wmma_f32_16x16x32_bf16(
              false, a[ti].v, false, b[0][tj].v, (short)0, accB[ti][tj], false, false);
          accC[ti][tj] = __builtin_amdgcn_wmma_f32_16x16x32_bf16(
              false, a[ti].v, false, b[1][tj].v, (short)0, accC[ti][tj], false, false);
          accX[ti][tj] = __builtin_amdgcn_wmma_f32_16x16x32_bf16(
              false, a[ti].v, false, b[2][tj].v, (short)0, accX[ti][tj], false, false);
        }
    }
    __syncthreads();                    // reads done before buf is overwritten
  }

  // ---- epilogue: bx = b*x (f32), c (bf16), new_states rows
#pragma unroll
  for (int ti = 0; ti < 2; ++ti)
#pragma unroll
    for (int tj = 0; tj < 2; ++tj) {
      v8f vbx = accB[ti][tj] * accX[ti][tj];
      v8f vc  = accC[ti][tj];
      const int gcol = j0 + wn * 32 + tj * 16 + lm;
#pragma unroll
      for (int r = 0; r < 8; ++r) {
        const int grow = m0 + wm * 32 + ti * 16 + r + 8 * lhi;
        const size_t idx = (size_t)grow * DCONV + gcol;
        bx[idx] = vbx[r];
        cg[idx] = f2bf(vc[r]);
        const int l = grow & (SEQLEN - 1);
        if (l >= SEQLEN - (KCONV - 1)) {
          const int seq = grow >> 12;
          states[(size_t)(seq * (KCONV - 1) + (l - (SEQLEN - KCONV + 1))) * DCONV
                 + gcol] = vbx[r];
        }
      }
    }
}

// ---------------------------------------------------------------------------
// Causal depthwise conv (K=4, zero history at sequence starts) + BCx gate.
// ---------------------------------------------------------------------------
__global__ void conv_gate(const float* __restrict__ bx,
                          const unsigned short* __restrict__ cg,
                          const float* __restrict__ w,      // [DCONV,1,KCONV]
                          unsigned short* __restrict__ y) {
  const int d  = blockIdx.x * blockDim.x + threadIdx.x;
  const int r0 = blockIdx.y * 16;
  const float w0 = w[d * KCONV + 0];
  const float w1 = w[d * KCONV + 1];
  const float w2 = w[d * KCONV + 2];
  const float w3 = w[d * KCONV + 3];

  float xm3 = 0.f, xm2 = 0.f, xm1 = 0.f;
  if ((r0 & (SEQLEN - 1)) != 0) {        // strips never straddle a sequence
    xm3 = bx[(size_t)(r0 - 3) * DCONV + d];
    xm2 = bx[(size_t)(r0 - 2) * DCONV + d];
    xm1 = bx[(size_t)(r0 - 1) * DCONV + d];
  }
#pragma unroll
  for (int r = 0; r < 16; ++r) {
    const size_t idx = (size_t)(r0 + r) * DCONV + d;
    const float xc = bx[idx];
    const float conv = w0 * xm3 + w1 * xm2 + w2 * xm1 + w3 * xc;
    y[idx] = f2bf(bf2f(cg[idx]) * conv);
    xm3 = xm2; xm2 = xm1; xm1 = xc;
  }
}

// ---------------------------------------------------------------------------
// GEMM2: out = y @ w_out^T.  Same async double-buffered tiling.
// Dynamic LDS: 4 * TILEU ushorts = 73728 B.
//   tile layout in smem (units of TILEU):  A0, A1, B0, B1
// ---------------------------------------------------------------------------
__global__ __launch_bounds__(512, 1)
void gemm2_out(const unsigned short* __restrict__ Y,    // bf16 [N_TOK x DCONV]
               const unsigned short* __restrict__ Wo,   // bf16 [HID x DCONV]
               float* __restrict__ out)                 // f32 [N_TOK x HID]
{
  extern __shared__ unsigned short smem[];

  const int tid  = threadIdx.x;
  const int lane = tid & 31;
  const int wave = tid >> 5;
  const int wm   = wave >> 2;
  const int wn   = wave & 3;
  const int m0   = blockIdx.y * BM;
  const int h0   = blockIdx.x * BM;

  const int lrow = tid >> 2;
  const int cc   = (tid & 3) * 16;
  const int lhi  = lane >> 4;
  const int lm   = lane & 15;

  v8f acc[2][2];
#pragma unroll
  for (int i = 0; i < 2; ++i)
#pragma unroll
    for (int j = 0; j < 2; ++j) acc[i][j] = (v8f)(0.0f);

  const unsigned short* srcA = Y  + (size_t)(m0 + lrow) * DCONV + cc;
  const unsigned short* srcB = Wo + (size_t)(h0 + lrow) * DCONV + cc;

  auto issue = [&](int buf, int kk) {
    unsigned short* dA = smem + buf * TILEU + lrow * LSTR2 + cc;
    unsigned short* dB = smem + (2 + buf) * TILEU + lrow * LSTR2 + cc;
    async_b128(dA,     srcA + kk);
    async_b128(dA + 8, srcA + kk + 8);
    async_b128(dB,     srcB + kk);
    async_b128(dB + 8, srcB + kk + 8);
  };

  const int NIT = DCONV / BK2;          // 32
  issue(0, 0);

  for (int it = 0; it < NIT; ++it) {
    const int buf = it & 1;
    if (it + 1 < NIT) {
      issue(1 - buf, (it + 1) * BK2);
      S_WAIT_ASYNC(0x4);
    } else {
      S_WAIT_ASYNC(0x0);
    }
    __syncthreads();

#pragma unroll
    for (int ks = 0; ks < 2; ++ks) {
      const int ksub = ks * 32;
      FragBF16 a[2], b[2];
#pragma unroll
      for (int ti = 0; ti < 2; ++ti) {
        const unsigned short* p =
            smem + buf * TILEU + (wm * 32 + ti * 16 + lm) * LSTR2 + ksub;
        a[ti].q[0] = *(const uint4*)(p + 8 * lhi);
        a[ti].q[1] = *(const uint4*)(p + 8 * lhi + 16);
      }
#pragma unroll
      for (int tj = 0; tj < 2; ++tj) {
        const unsigned short* p =
            smem + (2 + buf) * TILEU +
            (wn * 32 + tj * 16 + lm) * LSTR2 + ksub;
        b[tj].q[0] = *(const uint4*)(p + 8 * lhi);
        b[tj].q[1] = *(const uint4*)(p + 8 * lhi + 16);
      }
#pragma unroll
      for (int ti = 0; ti < 2; ++ti)
#pragma unroll
        for (int tj = 0; tj < 2; ++tj)
          acc[ti][tj] = __builtin_amdgcn_wmma_f32_16x16x32_bf16(
              false, a[ti].v, false, b[tj].v, (short)0, acc[ti][tj], false, false);
    }
    __syncthreads();
  }

#pragma unroll
  for (int ti = 0; ti < 2; ++ti)
#pragma unroll
    for (int tj = 0; tj < 2; ++tj) {
      const int gcol = h0 + wn * 32 + tj * 16 + lm;
#pragma unroll
      for (int r = 0; r < 8; ++r) {
        const int grow = m0 + wm * 32 + ti * 16 + r + 8 * lhi;
        out[(size_t)grow * HID + gcol] = acc[ti][tj][r];
      }
    }
}

// ---------------------------------------------------------------------------
extern "C" void kernel_launch(void* const* d_in, const int* in_sizes, int n_in,
                              void* d_out, int out_size, void* d_ws, size_t ws_size,
                              hipStream_t stream) {
  const float* hidden = (const float*)d_in[0];   // [16384 x 2048]
  const float* w_in   = (const float*)d_in[1];   // [6144 x 2048]
  const float* conv_w = (const float*)d_in[2];   // [2048 x 1 x 4]
  const float* w_out  = (const float*)d_in[3];   // [2048 x 2048]

  float* out    = (float*)d_out;
  float* states = out + (size_t)N_TOK * HID;

  // workspace layout (bytes):
  //   [0, 64MB)        hidden_bf16  (later reused as y_bf16)
  //   [64MB, 88MB)     w_in_bf16
  //   [88MB, 96MB)     w_out_bf16
  //   [96MB, 224MB)    bx  (f32)
  //   [224MB, 288MB)   c   (bf16)
  char* ws = (char*)d_ws;
  unsigned short* hid_bf  = (unsigned short*)(ws);
  unsigned short* win_bf  = (unsigned short*)(ws + (size_t)67108864);
  unsigned short* wout_bf = (unsigned short*)(ws + (size_t)92274688);
  float*          bx      = (float*)        (ws + (size_t)100663296);
  unsigned short* cg      = (unsigned short*)(ws + (size_t)234881024);

  const int lds1 = 8 * TILEU * (int)sizeof(unsigned short);  // 147456 B
  const int lds2 = 4 * TILEU * (int)sizeof(unsigned short);  //  73728 B
  (void)hipFuncSetAttribute((const void*)&gemm1_bcx,
                            hipFuncAttributeMaxDynamicSharedMemorySize, lds1);
  (void)hipFuncSetAttribute((const void*)&gemm2_out,
                            hipFuncAttributeMaxDynamicSharedMemorySize, lds2);

  // 1) convert operands to bf16
  {
    int n4 = (N_TOK * HID) / 4;
    cvt4_f32_bf16<<<dim3(4096), dim3(256), 0, stream>>>((const float4*)hidden,
                                                        (ushort4*)hid_bf, n4);
    n4 = (3 * DCONV * HID) / 4;
    cvt4_f32_bf16<<<dim3(4096), dim3(256), 0, stream>>>((const float4*)w_in,
                                                        (ushort4*)win_bf, n4);
    n4 = (HID * DCONV) / 4;
    cvt4_f32_bf16<<<dim3(2048), dim3(256), 0, stream>>>((const float4*)w_out,
                                                        (ushort4*)wout_bf, n4);
  }

  // 2) fused in_proj (b,c,x) -> bx, c, new_states
  gemm1_bcx<<<dim3(DCONV / BM, N_TOK / BM), dim3(512), lds1, stream>>>(
      hid_bf, win_bf, bx, cg, states);

  // 3) depthwise causal conv + c-gate -> y (bf16), reusing hidden_bf16 buffer
  conv_gate<<<dim3(DCONV / 256, N_TOK / 16), dim3(256), 0, stream>>>(
      bx, cg, conv_w, hid_bf);

  // 4) out_proj
  gemm2_out<<<dim3(HID / BM, N_TOK / BM), dim3(512), lds2, stream>>>(
      hid_bf, wout_bf, out);
}
